// SimpleLSTM_16363825397893
// MI455X (gfx1250) — compile-verified, matrix operand order checked by reading, not run on hardware
//
#include <hip/hip_runtime.h>
#include <hip/hip_bf16.h>

// Problem constants (match reference)
#define Hh   1024
#define G4H  4096   // 4*H
#define Bb   64
#define Ss   512

typedef __attribute__((ext_vector_type(16))) __bf16       v16bf;
typedef __attribute__((ext_vector_type(8)))  float        v8f;
typedef __attribute__((ext_vector_type(8)))  unsigned int v8u;

// ---------------------------------------------------------------- helpers
__device__ __forceinline__ unsigned short f2bf(float f) {
  unsigned int u = __float_as_uint(f);
  unsigned int r = 0x7FFFu + ((u >> 16) & 1u);   // round-to-nearest-even
  return (unsigned short)((u + r) >> 16);
}

__device__ __forceinline__ float sigmoidf_(float x) {
  return 1.0f / (1.0f + __expf(-x));
}

// Load a 16x32 bf16 fragment (A layout; B uses the same layout because we
// store W^T / U^T row-major as N x K). ISA 7.12.2: lane L holds row L%16,
// dword (2xK) offsets {0,1,2,3, 8,9,10,11} (+4 dwords for lanes 16..31).
__device__ __forceinline__ v16bf load_frag16x32(const unsigned short* __restrict__ base,
                                                int row, int ld, int k0) {
  int lane = threadIdx.x & 31;
  const uint4* p = (const uint4*)(base + (size_t)(row + (lane & 15)) * ld
                                       + k0 + ((lane >> 4) << 3));
  uint4 lo = p[0];   // K +0..7   (relative to lane-half)
  uint4 hi = p[2];   // K +16..23
  v8u d = { lo.x, lo.y, lo.z, lo.w, hi.x, hi.y, hi.z, hi.w };
  return __builtin_bit_cast(v16bf, d);
}

#define WMMA_BF16(A, B, C) \
  __builtin_amdgcn_wmma_f32_16x16x32_bf16(false, (A), false, (B), (short)0, (C), false, false)

// ---------------------------------------------------------------- converts
__global__ void convert_bf16_kernel(const float* __restrict__ src,
                                    unsigned short* __restrict__ dst, size_t n) {
  for (size_t i = (size_t)blockIdx.x * blockDim.x + threadIdx.x; i < n;
       i += (size_t)gridDim.x * blockDim.x)
    dst[i] = f2bf(src[i]);
}

// src (rows x cols) fp32 -> dst (cols x rows) bf16   (transpose + convert)
__global__ void transpose_bf16_kernel(const float* __restrict__ src,
                                      unsigned short* __restrict__ dst,
                                      int rows, int cols) {
  size_t total = (size_t)rows * cols;
  for (size_t i = (size_t)blockIdx.x * blockDim.x + threadIdx.x; i < total;
       i += (size_t)gridDim.x * blockDim.x) {
    size_t n = i / rows, k = i % rows;          // dst[n*rows + k]
    dst[i] = f2bf(src[k * (size_t)cols + n]);
  }
}

__global__ void init_state_kernel(const float* __restrict__ h0,
                                  const float* __restrict__ c0,
                                  float* __restrict__ cbuf,
                                  unsigned short* __restrict__ hbf, size_t n) {
  for (size_t i = (size_t)blockIdx.x * blockDim.x + threadIdx.x; i < n;
       i += (size_t)gridDim.x * blockDim.x) {
    cbuf[i] = c0[i];
    hbf[i]  = f2bf(h0[i]);
  }
}

// ---------------------------------------------------------------- pre-GEMM
// Gx(B*S x 4H) = Xb(B*S x H) @ W(H x 4H) + bias   (A bf16, B = W^T bf16, acc fp32)
// block = 256 threads = 8 waves; wave -> one M-tile x 4 N-tiles.
// This phase is occupancy-rich (131k waves); bounded by Gx traffic, not latency.
__global__ __launch_bounds__(256)
void gemm_xw_kernel(const unsigned short* __restrict__ Xb,
                    const unsigned short* __restrict__ WT,   // (4H x H) bf16
                    const float* __restrict__ bias,
                    float* __restrict__ Gx) {
  int wave  = threadIdx.x >> 5;
  int lane  = threadIdx.x & 31;
  int mrow  = blockIdx.x * 16;                         // 2048 M-tiles
  int nt0   = (blockIdx.y * 8 + wave) * 4;             // 4 consecutive N-tiles

  v8f c0 = {}, c1 = {}, c2 = {}, c3 = {};

  v16bf a  = load_frag16x32(Xb, mrow,       Hh, 0);
  v16bf b0 = load_frag16x32(WT, (nt0+0)*16, Hh, 0);
  v16bf b1 = load_frag16x32(WT, (nt0+1)*16, Hh, 0);
  v16bf b2 = load_frag16x32(WT, (nt0+2)*16, Hh, 0);
  v16bf b3 = load_frag16x32(WT, (nt0+3)*16, Hh, 0);

#pragma unroll 4
  for (int k0 = 32; k0 < Hh; k0 += 32) {
    v16bf an  = load_frag16x32(Xb, mrow,       Hh, k0);
    v16bf b0n = load_frag16x32(WT, (nt0+0)*16, Hh, k0);
    v16bf b1n = load_frag16x32(WT, (nt0+1)*16, Hh, k0);
    v16bf b2n = load_frag16x32(WT, (nt0+2)*16, Hh, k0);
    v16bf b3n = load_frag16x32(WT, (nt0+3)*16, Hh, k0);
    c0 = WMMA_BF16(a, b0, c0);
    c1 = WMMA_BF16(a, b1, c1);
    c2 = WMMA_BF16(a, b2, c2);
    c3 = WMMA_BF16(a, b3, c3);
    a = an; b0 = b0n; b1 = b1n; b2 = b2n; b3 = b3n;
  }
  c0 = WMMA_BF16(a, b0, c0);
  c1 = WMMA_BF16(a, b1, c1);
  c2 = WMMA_BF16(a, b2, c2);
  c3 = WMMA_BF16(a, b3, c3);

  int mb = (lane >> 4) << 3;     // +8 rows for lanes 16..31
  int n  = lane & 15;
#pragma unroll
  for (int v = 0; v < 8; ++v) {
    size_t m = (size_t)(mrow + mb + v);
    Gx[m * G4H + (nt0+0)*16 + n] = c0[v] + bias[(nt0+0)*16 + n];
    Gx[m * G4H + (nt0+1)*16 + n] = c1[v] + bias[(nt0+1)*16 + n];
    Gx[m * G4H + (nt0+2)*16 + n] = c2[v] + bias[(nt0+2)*16 + n];
    Gx[m * G4H + (nt0+3)*16 + n] = c3[v] + bias[(nt0+3)*16 + n];
  }
}

// ---------------------------------------------------------------- LSTM step
// Latency-critical phase. Split-K=4 shortens the dependent chain to 8 WMMAs
// per wave, and ALL 16 fragments (32 x global_load_b128) are issued as one
// burst into distinct register arrays, so one L2 round-trip latency is paid
// once per wave; the WMMA chain then drains with partial loadcnt waits.
// Block = 512 threads = 16 waves: wave w -> gate (w&3), K-quarter (w>>2).
// Partials reduced through LDS inside the fused sigmoid/tanh cell update.
// grid = (4 batch-tiles, 64 hidden-slices).
__global__ __launch_bounds__(512)
void lstm_step_kernel(const unsigned short* __restrict__ hIn,   // (B x H) bf16, step t-1
                      const unsigned short* __restrict__ UT,    // (4H x H) bf16
                      const float* __restrict__ Gx,             // (B*S x 4H)
                      float* __restrict__ cbuf,                 // (B x H) fp32, in/out
                      unsigned short* __restrict__ hOut,        // (B x H) bf16, step t
                      float* __restrict__ out_seq,              // (B,S,H)
                      float* __restrict__ out_h,
                      float* __restrict__ out_c,
                      int t) {
  int wave = threadIdx.x >> 5;
  int gate = wave & 3;                    // i,f,g,o
  int kq   = wave >> 2;                   // K-quarter 0..3
  int lane = threadIdx.x & 31;
  int mrow = blockIdx.x * 16;             // batch tile
  int jcol = blockIdx.y * 16;             // hidden slice
  int ncol = gate * Hh + jcol;            // column in the 4H gate space
  int kb   = kq * (Hh / 4);               // this wave's K slice base

  // burst-load all fragments (kept in registers; fully unrolled)
  v16bf ar[8], br[8];
#pragma unroll
  for (int i = 0; i < 8; ++i) {
    ar[i] = load_frag16x32(hIn, mrow, Hh, kb + 32 * i);
    br[i] = load_frag16x32(UT,  ncol, Hh, kb + 32 * i);
  }

  v8f acc = {};
#pragma unroll
  for (int i = 0; i < 8; ++i)
    acc = WMMA_BF16(ar[i], br[i], acc);

  __shared__ float gl[16][16][16];        // [wave = kq*4+gate][m][n]
  int mb = (lane >> 4) << 3;
  int n  = lane & 15;
#pragma unroll
  for (int v = 0; v < 8; ++v)
    gl[wave][mb + v][n] = acc[v];
  __syncthreads();

  // fused cell update: threads 0..255 handle the 16x16 (batch x hidden) slab
  int e = threadIdx.x;
  if (e < 256) {
    int m = e >> 4, nn = e & 15;
    size_t brow = (size_t)(mrow + m);
    int    hcol = jcol + nn;
    const float* gx = Gx + (brow * Ss + t) * G4H + jcol + nn;
    float ig = gl[0][m][nn] + gl[4][m][nn] + gl[ 8][m][nn] + gl[12][m][nn] + gx[0 * Hh];
    float fg = gl[1][m][nn] + gl[5][m][nn] + gl[ 9][m][nn] + gl[13][m][nn] + gx[1 * Hh];
    float gg = gl[2][m][nn] + gl[6][m][nn] + gl[10][m][nn] + gl[14][m][nn] + gx[2 * Hh];
    float og = gl[3][m][nn] + gl[7][m][nn] + gl[11][m][nn] + gl[15][m][nn] + gx[3 * Hh];
    float i_t = sigmoidf_(ig);
    float f_t = sigmoidf_(fg);
    float g_t = tanhf(gg);
    float o_t = sigmoidf_(og);
    size_t cidx = brow * Hh + hcol;
    float c_new = f_t * cbuf[cidx] + i_t * g_t;
    cbuf[cidx] = c_new;
    float h_new = o_t * tanhf(c_new);
    out_seq[(brow * Ss + t) * Hh + hcol] = h_new;
    hOut[cidx] = f2bf(h_new);
    if (t == Ss - 1) { out_h[cidx] = h_new; out_c[cidx] = c_new; }
  }
}

// ---------------------------------------------------------------- launch
extern "C" void kernel_launch(void* const* d_in, const int* in_sizes, int n_in,
                              void* d_out, int out_size, void* d_ws, size_t ws_size,
                              hipStream_t stream) {
  const float* x    = (const float*)d_in[0];   // (B,S,H)
  const float* h0   = (const float*)d_in[1];   // (B,H)
  const float* c0   = (const float*)d_in[2];   // (B,H)
  const float* W    = (const float*)d_in[3];   // (H,4H)
  const float* U    = (const float*)d_in[4];   // (H,4H)
  const float* bias = (const float*)d_in[5];   // (4H)

  float* out      = (float*)d_out;
  float* out_seq  = out;                                   // (B,S,H)
  float* out_h    = out + (size_t)Bb * Ss * Hh;            // (B,H)
  float* out_c    = out_h + (size_t)Bb * Hh;               // (B,H)

  // workspace carve-up (all chunks 256B-aligned sizes)
  char* ws = (char*)d_ws;
  float*          Gx   = (float*)ws;          ws += (size_t)Bb * Ss * G4H * sizeof(float);   // 512 MB
  unsigned short* WT   = (unsigned short*)ws; ws += (size_t)G4H * Hh * sizeof(unsigned short);
  unsigned short* UT   = (unsigned short*)ws; ws += (size_t)G4H * Hh * sizeof(unsigned short);
  unsigned short* Xb   = (unsigned short*)ws; ws += (size_t)Bb * Ss * Hh * sizeof(unsigned short);
  float*          cbuf = (float*)ws;          ws += (size_t)Bb * Hh * sizeof(float);
  unsigned short* hbf0 = (unsigned short*)ws; ws += (size_t)Bb * Hh * sizeof(unsigned short);
  unsigned short* hbf1 = (unsigned short*)ws; ws += (size_t)Bb * Hh * sizeof(unsigned short);

  // 1) precision conversion / transposes (one-time, bandwidth-trivial)
  transpose_bf16_kernel<<<8192, 256, 0, stream>>>(W, WT, Hh, G4H);
  transpose_bf16_kernel<<<8192, 256, 0, stream>>>(U, UT, Hh, G4H);
  convert_bf16_kernel<<<16384, 256, 0, stream>>>(x, Xb, (size_t)Bb * Ss * Hh);
  init_state_kernel<<<256, 256, 0, stream>>>(h0, c0, cbuf, hbf0, (size_t)Bb * Hh);

  // 2) hoisted input GEMM: Gx = X @ W + bias  (275 GFLOP, fully parallel)
  gemm_xw_kernel<<<dim3(2048, 8), 256, 0, stream>>>(Xb, WT, bias, Gx);

  // 3) sequential recurrence; per-step launch is the inter-step barrier.
  //    h double-buffered (blocks of step t must not see step-t writes).
  for (int t = 0; t < Ss; ++t) {
    const unsigned short* hin = (t & 1) ? hbf1 : hbf0;
    unsigned short*       hot = (t & 1) ? hbf0 : hbf1;
    lstm_step_kernel<<<dim3(4, 64), 512, 0, stream>>>(hin, UT, Gx, cbuf, hot,
                                                      out_seq, out_h, out_c, t);
  }
}
